// NSMMPP_50414326121065
// MI455X (gfx1250) — matrix operand order repaired and checked by reading, not compiled, not run on hardware
//
#include <hip/hip_runtime.h>

// ---------------- types for WMMA ----------------
typedef __bf16 bf16_t;
typedef __attribute__((ext_vector_type(16))) __bf16 v16bf;
typedef __attribute__((ext_vector_type(8)))  __bf16 v8bf;
typedef __attribute__((ext_vector_type(8)))  float  v8f;

#define H_DIM   1024
#define N_EV    4096
#define N_STEP  4095      // rows of Wx / scan steps
#define N_SIM   4096
#define J_DIM   7168      // 7 * H
#define TGT     1
#define G_REC   64        // persistent workgroups in recurrence

// ---------------- small math helpers ----------------
__device__ __forceinline__ float sigmoidf_(float x) { return 1.0f / (1.0f + expf(-x)); }
__device__ __forceinline__ float softplusf_(float x) {
  // numerically stable log1p(exp(x))
  return x > 0.0f ? x + log1pf(expf(-x)) : log1pf(expf(x));
}
__device__ __forceinline__ float block_reduce_256(float v, float* sm) {
  int t = threadIdx.x;
  sm[t] = v; __syncthreads();
  for (int s = 128; s > 0; s >>= 1) {
    if (t < s) sm[t] += sm[t + s];
    __syncthreads();
  }
  return sm[0];
}

// ---------------- init: zero h-state + barrier ----------------
__global__ void init_kernel(float* g_h, unsigned* bar) {
  int t = threadIdx.x;
  if (t < H_DIM) g_h[t] = 0.0f;
  if (t == 0) *bar = 0u;
}

// ---------------- convert W (7,H,H) fp32 -> bf16 flat [j][k] ----------------
__global__ __launch_bounds__(256) void convert_w_kernel(const float* __restrict__ W,
                                                        bf16_t* __restrict__ wb) {
  int idx = blockIdx.x * 256 + threadIdx.x;
  if (idx < 7 * H_DIM * H_DIM) wb[idx] = (bf16_t)W[idx];
}

// ---------------- transpose U: Ut[e][k][h] = U[e][h][k] (fp32, LDS tiled) ----
__global__ __launch_bounds__(256) void transpose_u_kernel(const float* __restrict__ U,
                                                          float* __restrict__ Ut) {
  __shared__ float tile[32][33];
  int e  = blockIdx.z;
  int hb = blockIdx.y * 32, kb = blockIdx.x * 32;
  int tx = threadIdx.x, ty = threadIdx.y;              // 32 x 8
  const float* src = U  + ((size_t)e << 20);
  float*       dst = Ut + ((size_t)e << 20);
  #pragma unroll
  for (int i = 0; i < 32; i += 8)
    tile[ty + i][tx] = src[(size_t)(hb + ty + i) * H_DIM + kb + tx];
  __syncthreads();
  #pragma unroll
  for (int i = 0; i < 32; i += 8)
    dst[(size_t)(kb + ty + i) * H_DIM + hb + tx] = tile[tx][ty + i];
}

// ---------------- gather embedding rows -> bf16, pad row 4095 with 0 --------
__global__ __launch_bounds__(256) void gather_emb_kernel(const float* __restrict__ Emb,
                                                         const int* __restrict__ label_seq,
                                                         bf16_t* __restrict__ emb) {
  int idx = blockIdx.x * 256 + threadIdx.x;
  if (idx >= N_EV * H_DIM) return;
  int n = idx >> 10, k = idx & (H_DIM - 1);
  float v = (n < N_STEP) ? Emb[k * 33 + label_seq[n]] : 0.0f;
  emb[idx] = (bf16_t)v;
}

// ---------------- Wx GEMM via v_wmma_f32_16x16x32_bf16 ----------------------
// Wx[n][j] = sum_k emb[n][k] * Wb[j][k]   (M=4096 padded, N=7168, K=1024)
__global__ __launch_bounds__(256) void wx_wmma_kernel(const bf16_t* __restrict__ emb,
                                                      const bf16_t* __restrict__ wb,
                                                      float* __restrict__ wx) {
  int wid  = (blockIdx.x * 256 + threadIdx.x) >> 5;    // global wave id
  int lane = threadIdx.x & 31;
  int mt = wid / (J_DIM / 16);                         // 0..255
  int nt = wid % (J_DIM / 16);                         // 0..447
  int row  = lane & 15;
  int half = lane >> 4;

  // A (16x32 bf16): lane<16 -> K 0..15; lane>=16 -> K 16..31
  const bf16_t* pa = emb + (size_t)(mt * 16 + row) * H_DIM + half * 16;
  // B (32x16): lane = column; lanes<16 -> K 0..15, lanes>=16 -> K 16..31
  const bf16_t* pb = wb  + (size_t)(nt * 16 + row) * H_DIM + half * 16;

  v8f acc = {};
  for (int kk = 0; kk < H_DIM / 32; ++kk) {
    v8bf a0 = *(const v8bf*)(pa);
    v8bf a1 = *(const v8bf*)(pa + 8);
    v8bf b0 = *(const v8bf*)(pb);
    v8bf b1 = *(const v8bf*)(pb + 8);
    v16bf a = __builtin_shufflevector(a0, a1, 0,1,2,3,4,5,6,7,8,9,10,11,12,13,14,15);
    v16bf b = __builtin_shufflevector(b0, b1, 0,1,2,3,4,5,6,7,8,9,10,11,12,13,14,15);
    acc = __builtin_amdgcn_wmma_f32_16x16x32_bf16(false, a, false, b,
                                                  (short)0, acc, false, false);
    pa += 32; pb += 32;
  }
  // D: lanes<16 rows 0..7, lanes>=16 rows 8..15; col = lane&15
  int col  = nt * 16 + (lane & 15);
  int row0 = mt * 16 + half * 8;
  #pragma unroll
  for (int r = 0; r < 8; ++r)
    wx[(size_t)(row0 + r) * J_DIM + col] = acc[r];
}

// ---------------- persistent recurrence: one grid-sync per step -------------
// WG g owns h-columns [g*16, g*16+16) for ALL 7 gates; U kept fp32 in L2.
__global__ __launch_bounds__(256) void recurrence_kernel(
    const float* __restrict__ ut,      // [7][1024][1024]  Ut[e][k][h]
    const float* __restrict__ wx,      // [4096][7168]
    const float* __restrict__ dbias,   // [7][1024]
    const float* __restrict__ time_seq,
    float* g_h,                        // [1024] broadcast h
    unsigned* bar,
    float* __restrict__ allh,  float* __restrict__ allc,
    float* __restrict__ allcb, float* __restrict__ alldel,
    float* __restrict__ allo) {
  __shared__ float h_sh[H_DIM];
  __shared__ float part_sh[224];
  __shared__ float temp_sh[112];
  const int g  = blockIdx.x;       // 0..63
  const int t  = threadIdx.x;      // 0..255
  const int h0 = g * 16;
  const int prt = t / 112;         // k-half (t<224)
  const int out = t % 112;         // e*16 + hl
  const int e  = out >> 4;
  const int hl = out & 15;

  float c_st = 0.0f, cb_st = 0.0f;

  for (int n = 0; n < N_STEP; ++n) {
    // broadcast h (L2-resident, 4KB)
    for (int k = t; k < H_DIM; k += 256) h_sh[k] = g_h[k];
    __syncthreads();

    if (t < 224) {
      const float* up = ut + ((size_t)e << 20) + (size_t)(prt * 512) * H_DIM + h0 + hl;
      float acc = 0.0f;
      #pragma unroll 8
      for (int k = 0; k < 512; ++k)
        acc = fmaf(up[(size_t)k * H_DIM], h_sh[prt * 512 + k], acc);
      part_sh[t] = acc;
    }
    __syncthreads();

    if (t < 112) {
      temp_sh[out] = part_sh[t] + part_sh[t + 112]
                   + wx[(size_t)n * J_DIM + (e << 10) + h0 + hl]
                   + dbias[(e << 10) + h0 + hl];
    }
    __syncthreads();

    if (t < 16) {
      float T0 = temp_sh[0*16+t], T1 = temp_sh[1*16+t], T2 = temp_sh[2*16+t],
            T3 = temp_sh[3*16+t], T4 = temp_sh[4*16+t], T5 = temp_sh[5*16+t],
            T6 = temp_sh[6*16+t];
      float ig = sigmoidf_(T0), fg = sigmoidf_(T1), z = tanhf(T2), o = sigmoidf_(T3);
      float ib = sigmoidf_(T4), fb = sigmoidf_(T5), del = softplusf_(T6);
      float c  = fg * c_st  + ig * z;
      float cb = fb * cb_st + ib * z;
      float dt = time_seq[n + 1] - time_seq[n];
      float ct = cb + (c - cb) * expf(-del * dt);
      float hn = o * tanhf(ct);
      c_st = ct; cb_st = cb;
      size_t off = (size_t)n * H_DIM + h0 + t;
      allh[off] = hn; allc[off] = c; allcb[off] = cb; alldel[off] = del; allo[off] = o;
      g_h[h0 + t] = hn;
    }
    __threadfence();                 // release: make g_h visible device-wide
    __syncthreads();

    if (t == 0) {                    // grid-wide barrier (generation counted)
      unsigned target = (unsigned)(n + 1) * G_REC;
      atomicAdd(bar, 1u);
      while (atomicAdd(bar, 0u) < target) __builtin_amdgcn_s_sleep(2);
    }
    __syncthreads();
    __threadfence();                 // acquire: see other WGs' g_h writes
  }
}

// ---------------- term1: masked log(lambda) per event row -------------------
__global__ __launch_bounds__(256) void term1_kernel(const float* __restrict__ allh,
                                                    const float* __restrict__ w,
                                                    const float* __restrict__ log_s,
                                                    const int* __restrict__ label_seq,
                                                    const int* __restrict__ ignore_first,
                                                    float* __restrict__ t1) {
  __shared__ float sm[256];
  int r = blockIdx.x;                          // 0..4093 (all_h row index)
  int beg = (*ignore_first) ? 1 : 0;
  bool active = (r >= beg) && (label_seq[r + 1] == TGT);
  if (!active) {                               // uniform across the block
    if (threadIdx.x == 0) t1[r] = 0.0f;
    return;
  }
  float acc = 0.0f;
  for (int k = threadIdx.x; k < H_DIM; k += 256)
    acc = fmaf(allh[(size_t)r * H_DIM + k], w[k], acc);   // w[0][:] (TGT-1 == 0)
  float dot = block_reduce_256(acc, sm);
  if (threadIdx.x == 0) {
    float s0  = expf(log_s[TGT - 1]);
    float lam = s0 * softplusf_(dot / s0) + 1e-9f;
    t1[r] = logf(lam);
  }
}

// ---------------- term2: lambda at simulated times --------------------------
__global__ __launch_bounds__(256) void term2_kernel(const float* __restrict__ allc,
                                                    const float* __restrict__ allcb,
                                                    const float* __restrict__ alldel,
                                                    const float* __restrict__ allo,
                                                    const int* __restrict__ sim_time_idx,
                                                    const float* __restrict__ sim_time_seq,
                                                    const float* __restrict__ time_seq,
                                                    const float* __restrict__ w,
                                                    const float* __restrict__ log_s,
                                                    float* __restrict__ t2) {
  __shared__ float sm[256];
  int s = blockIdx.x;                          // 0..4095
  int i = sim_time_idx[s];
  float dtv = sim_time_seq[s] - time_seq[i];
  float acc = 0.0f;
  for (int k = threadIdx.x; k < H_DIM; k += 256) {
    size_t off = (size_t)i * H_DIM + k;
    float cb   = allcb[off];
    float csim = cb + (allc[off] - cb) * expf(-alldel[off] * dtv);
    float hsim = allo[off] * tanhf(csim);
    acc = fmaf(hsim, w[k], acc);
  }
  float dot = block_reduce_256(acc, sm);
  if (threadIdx.x == 0) {
    float s0 = expf(log_s[TGT - 1]);
    t2[s] = s0 * softplusf_(dot / s0) + 1e-9f;
  }
}

// ---------------- deterministic final reduction -----------------------------
__global__ __launch_bounds__(256) void final_kernel(const float* __restrict__ t1,
                                                    const float* __restrict__ t2,
                                                    const float* __restrict__ time_seq,
                                                    float* __restrict__ out) {
  __shared__ float sm[256];
  float a = 0.0f, b = 0.0f;
  for (int i = threadIdx.x; i < N_STEP - 1; i += 256) a += t1[i];   // rows 0..4093
  for (int i = threadIdx.x; i < N_SIM; i += 256)      b += t2[i];
  float term1 = block_reduce_256(a, sm);
  __syncthreads();
  float term2m = block_reduce_256(b, sm);
  if (threadIdx.x == 0) {
    float span  = time_seq[N_EV - 1] - time_seq[0];
    float term2 = (term2m / (float)N_SIM) * span;
    out[0] = -(term1 - term2);
  }
}

// ---------------- host: workspace carving + launches ------------------------
extern "C" void kernel_launch(void* const* d_in, const int* in_sizes, int n_in,
                              void* d_out, int out_size, void* d_ws, size_t ws_size,
                              hipStream_t stream) {
  const int*   label_seq    = (const int*)  d_in[0];
  const float* time_seq     = (const float*)d_in[1];
  const float* sim_time_seq = (const float*)d_in[2];
  const int*   sim_time_idx = (const int*)  d_in[3];
  const int*   ignore_first = (const int*)  d_in[4];
  const float* Emb          = (const float*)d_in[5];
  const float* W            = (const float*)d_in[6];
  const float* U            = (const float*)d_in[7];
  const float* dbias        = (const float*)d_in[8];
  const float* w            = (const float*)d_in[9];
  const float* log_s        = (const float*)d_in[10];
  float* out = (float*)d_out;

  char* ws = (char*)d_ws;
  size_t off = 0;
  auto take = [&](size_t bytes) -> char* {
    char* p = ws + off;
    off = (off + bytes + 255) & ~(size_t)255;
    return p;
  };
  bf16_t*   wb     = (bf16_t*)  take((size_t)J_DIM * H_DIM * 2);
  float*    ut     = (float*)   take((size_t)7 * H_DIM * H_DIM * 4);
  bf16_t*   emb    = (bf16_t*)  take((size_t)N_EV * H_DIM * 2);
  float*    wx     = (float*)   take((size_t)N_EV * J_DIM * 4);
  float*    allh   = (float*)   take((size_t)N_STEP * H_DIM * 4);
  float*    allc   = (float*)   take((size_t)N_STEP * H_DIM * 4);
  float*    allcb  = (float*)   take((size_t)N_STEP * H_DIM * 4);
  float*    alldel = (float*)   take((size_t)N_STEP * H_DIM * 4);
  float*    allo   = (float*)   take((size_t)N_STEP * H_DIM * 4);
  float*    g_h    = (float*)   take(H_DIM * 4);
  unsigned* bar    = (unsigned*)take(256);
  float*    t1     = (float*)   take((size_t)N_STEP * 4);
  float*    t2     = (float*)   take((size_t)N_SIM * 4);

  init_kernel<<<1, 1024, 0, stream>>>(g_h, bar);
  convert_w_kernel<<<(7 * H_DIM * H_DIM) / 256, 256, 0, stream>>>(W, wb);
  transpose_u_kernel<<<dim3(32, 32, 7), dim3(32, 8), 0, stream>>>(U, ut);
  gather_emb_kernel<<<(N_EV * H_DIM) / 256, 256, 0, stream>>>(Emb, label_seq, emb);
  wx_wmma_kernel<<<(N_EV / 16) * (J_DIM / 16) / 8, 256, 0, stream>>>(emb, wb, wx);
  recurrence_kernel<<<G_REC, 256, 0, stream>>>(ut, wx, dbias, time_seq, g_h, bar,
                                               allh, allc, allcb, alldel, allo);
  term1_kernel<<<N_STEP - 1, 256, 0, stream>>>(allh, w, log_s, label_seq, ignore_first, t1);
  term2_kernel<<<N_SIM, 256, 0, stream>>>(allc, allcb, alldel, allo,
                                          sim_time_idx, sim_time_seq, time_seq, w, log_s, t2);
  final_kernel<<<1, 256, 0, stream>>>(t1, t2, time_seq, out);
}
